// RelationNet_53850299957574
// MI455X (gfx1250) — matrix operand order repaired and checked by reading, not compiled, not run on hardware
//
#include <hip/hip_runtime.h>
#include <hip/hip_bf16.h>
#include <cfloat>
#include <math.h>

typedef __bf16 bf16;
typedef bf16  v8bf  __attribute__((ext_vector_type(8)));
typedef bf16  v16bf __attribute__((ext_vector_type(16)));
typedef float v8f   __attribute__((ext_vector_type(8)));

#define BSZ 16
#define NN  1000
#define KK  10
#define DD  256
#define F3  576
#define ROWS  (BSZ*NN)      // 16000
#define MROWS (ROWS*KK)     // 160000
#define MB    32            // M rows per workgroup for GEMM kernels (2 x 16-row tiles)
#define LDA2  264           // bf16 LDS stride for 256-wide panels (16B-aligned rows)
#define LDF2  584           // bf16 LDS stride for 576-wide feature panel

// ---------- WMMA bf16 16x16x32, f32 accumulate ----------
static __device__ inline v8f wmma_bf16(v16bf a, v16bf b, v8f c) {
    return __builtin_amdgcn_wmma_f32_16x16x32_bf16(false, a, false, b, (short)0, c, false, false);
}

// A fragment (16x32 bf16) from an LDS row: two contiguous 16B chunks per lane.
// lanes 0-15: K = k+0..7 and k+16..23 ; lanes 16-31: K = k+8..15 and k+24..31
static __device__ inline v16bf load_afrag(const bf16* rowptr, int k, int h) {
    v8bf lo = *(const v8bf*)(rowptr + k + 8*h);
    v8bf hi = *(const v8bf*)(rowptr + k + 16 + 8*h);
    return __builtin_shufflevector(lo, hi, 0,1,2,3,4,5,6,7,8,9,10,11,12,13,14,15);
}

// packed B fragment: contiguous 16 bf16 (32B) per lane
static __device__ inline v16bf load_bfrag(const bf16* wp, int kt, int nt, int lane) {
    return *(const v16bf*)(wp + (((size_t)kt*16 + nt)*32 + lane)*16);
}

// sign-aware float atomic max (init -FLT_MAX)
static __device__ inline void atomicMaxFloat(float* addr, float val) {
    if (val >= 0.0f) atomicMax((int*)addr, __float_as_int(val));
    else             atomicMin((unsigned int*)addr, __float_as_uint(val));
}

// ---------- prep: pack f32 weight [Kdim][256] into per-lane bf16 B-fragment layout ----------
// flat packed index t: e=t&15, lane=(t>>4)&31, nt=(t>>9)&15, kt=t>>13
// K = kt*32 + 8*(lane>>4) + (e&7) + 16*(e>>3) ; N = nt*16 + (lane&15)
__global__ void pack_weight_kernel(const float* __restrict__ w, bf16* __restrict__ wp, int kdim) {
    int t = blockIdx.x * 256 + threadIdx.x;
    if (t >= kdim * 256) return;
    int e    = t & 15;
    int lane = (t >> 4) & 31;
    int nt   = (t >> 9) & 15;
    int kt   = t >> 13;
    int Kr   = kt*32 + 8*(lane >> 4) + (e & 7) + 16*(e >> 3);
    int n    = nt*16 + (lane & 15);
    wp[t] = (bf16)w[(size_t)Kr*DD + n];
}

// ---------- K0: init agg to -FLT_MAX ----------
__global__ void init_agg_kernel(float* agg, int n) {
    int i = blockIdx.x * 256 + threadIdx.x;
    if (i < n) agg[i] = -FLT_MAX;
}

// ---------- K1: IoU + attn_mask + top-K(10) ----------
__global__ __launch_bounds__(32) void topk_iou_kernel(const float* __restrict__ pred_boxes,
                                                      const float* __restrict__ seed_mask,
                                                      float* __restrict__ attn_out,
                                                      float* __restrict__ tv,
                                                      int*   __restrict__ idx,
                                                      float* __restrict__ mk) {
    const int blk = blockIdx.x;         // 0..15999
    const int b = blk / NN, i = blk % NN;
    const int lane = threadIdx.x;

    const float* pb = pred_boxes + (size_t)b * NN * 4;
    float cx = pb[i*4+0], cy = pb[i*4+1], w = pb[i*4+2], h = pb[i*4+3];
    float ix0 = cx - 0.5f*w, iy0 = cy - 0.5f*h, ix1 = cx + 0.5f*w, iy1 = cy + 0.5f*h;
    float areai = (ix1-ix0)*(iy1-iy0);
    float negi  = 1.0f - seed_mask[b*NN + i];

    float lv[KK]; int li[KK];
#pragma unroll
    for (int q = 0; q < KK; q++) { lv[q] = -1.0f; li[q] = 0x7fffffff; }

    float* amrow = attn_out + (size_t)b*NN*NN + (size_t)i*NN;
    for (int j = lane; j < NN; j += 32) {
        float jcx = pb[j*4+0], jcy = pb[j*4+1], jw = pb[j*4+2], jh = pb[j*4+3];
        float jx0 = jcx-0.5f*jw, jy0 = jcy-0.5f*jh, jx1 = jcx+0.5f*jw, jy1 = jcy+0.5f*jh;
        float areaj = (jx1-jx0)*(jy1-jy0);
        float lx = fmaxf(ix0, jx0), ly = fmaxf(iy0, jy0);
        float rx = fminf(ix1, jx1), ry = fminf(iy1, jy1);
        float iw = fmaxf(rx-lx, 0.0f), ih = fmaxf(ry-ly, 0.0f);
        float inter = iw*ih;
        float iou = inter / (areai + areaj - inter);
        amrow[j] = (iou >= 0.5f) ? 1.0f : 0.0f;
        float ov = iou * seed_mask[b*NN + j] * negi;
        float cv = ov; int ci = j;
#pragma unroll
        for (int q = 0; q < KK; q++) {
            bool better = (cv > lv[q]) || (cv == lv[q] && ci < li[q]);
            float ov2 = lv[q]; int oi2 = li[q];
            if (better) { lv[q] = cv; li[q] = ci; cv = ov2; ci = oi2; }
        }
    }

    __shared__ float sv[32*KK];
    __shared__ int   si[32*KK];
    __shared__ float rv[KK];
    __shared__ int   ri[KK];
#pragma unroll
    for (int q = 0; q < KK; q++) { sv[lane*KK+q] = lv[q]; si[lane*KK+q] = li[q]; }
    __syncthreads();
    if (lane == 0) {
        for (int q = 0; q < KK; q++) {
            float best = -2.0f; int bidx = 0x7fffffff; int bp = 0;
            for (int t = 0; t < 32*KK; t++) {
                float v = sv[t];
                if (v > best || (v == best && si[t] < bidx)) { best = v; bidx = si[t]; bp = t; }
            }
            rv[q] = best; ri[q] = bidx; sv[bp] = -2.0f;
        }
    }
    __syncthreads();
    if (lane < KK) {
        float v = rv[lane]; int j = ri[lane];
        float m = seed_mask[b*NN + j] * ((v >= 0.5f) ? 1.0f : 0.0f);
        size_t o = (size_t)blk*KK + lane;
        tv[o] = v; idx[o] = j; mk[o] = m;
    }
}

// ---------- K2: cur = relu(tgt@w1+b1)@w2+b2 (bf16 WMMA, 2x2 tile jobs) ----------
__global__ __launch_bounds__(256) void cur_kernel(const float* __restrict__ tgt,
                                                  const bf16* __restrict__ w1p, const float* __restrict__ b1,
                                                  const bf16* __restrict__ w2p, const float* __restrict__ b2,
                                                  float* __restrict__ cur) {
    __shared__ bf16 A[MB*LDA2];
    __shared__ bf16 X[MB*LDA2];
    const int blk = blockIdx.x;                 // 0..499
    const int tid = threadIdx.x, lane = tid & 31, wv = tid >> 5;
    const int h = lane >> 4, col = lane & 15;
    const int nt0 = wv*2, nt1 = nt0 + 1;        // each wave: 2 N-tiles x both M-tiles
    const int r0 = (lane & 15)*LDA2;            // A row for mt=0
    const int r1 = (16 + (lane & 15))*LDA2;     // A row for mt=1
    const int rb0 = h*8, rb1 = 16 + h*8;        // C row bases

    const float* Tg = tgt + (size_t)blk*MB*DD;
    for (int t = tid; t < MB*DD; t += 256) A[(t>>8)*LDA2 + (t&255)] = (bf16)Tg[t];
    __syncthreads();

    v8f a00 = {}, a01 = {}, a10 = {}, a11 = {};
    for (int kt = 0; kt < DD/32; kt++) {
        v16bf fa0 = load_afrag(&A[r0], kt*32, h);
        v16bf fa1 = load_afrag(&A[r1], kt*32, h);
        v16bf fb0 = load_bfrag(w1p, kt, nt0, lane);
        v16bf fb1 = load_bfrag(w1p, kt, nt1, lane);
        a00 = wmma_bf16(fa0, fb0, a00);
        a10 = wmma_bf16(fa1, fb0, a10);
        a01 = wmma_bf16(fa0, fb1, a01);
        a11 = wmma_bf16(fa1, fb1, a11);
    }
    {
        float bb0 = b1[nt0*16 + col], bb1 = b1[nt1*16 + col];
#pragma unroll
        for (int r = 0; r < 8; r++) {
            X[(rb0+r)*LDA2 + nt0*16 + col] = (bf16)fmaxf(a00[r] + bb0, 0.0f);
            X[(rb0+r)*LDA2 + nt1*16 + col] = (bf16)fmaxf(a01[r] + bb1, 0.0f);
            X[(rb1+r)*LDA2 + nt0*16 + col] = (bf16)fmaxf(a10[r] + bb0, 0.0f);
            X[(rb1+r)*LDA2 + nt1*16 + col] = (bf16)fmaxf(a11[r] + bb1, 0.0f);
        }
    }
    __syncthreads();

    a00 = (v8f){}; a01 = (v8f){}; a10 = (v8f){}; a11 = (v8f){};
    for (int kt = 0; kt < DD/32; kt++) {
        v16bf fa0 = load_afrag(&X[r0], kt*32, h);
        v16bf fa1 = load_afrag(&X[r1], kt*32, h);
        v16bf fb0 = load_bfrag(w2p, kt, nt0, lane);
        v16bf fb1 = load_bfrag(w2p, kt, nt1, lane);
        a00 = wmma_bf16(fa0, fb0, a00);
        a10 = wmma_bf16(fa1, fb0, a10);
        a01 = wmma_bf16(fa0, fb1, a01);
        a11 = wmma_bf16(fa1, fb1, a11);
    }
    {
        float bb0 = b2[nt0*16 + col], bb1 = b2[nt1*16 + col];
        float* Cg = cur + (size_t)blk*MB*DD;
#pragma unroll
        for (int r = 0; r < 8; r++) {
            Cg[(rb0+r)*DD + nt0*16 + col] = a00[r] + bb0;
            Cg[(rb0+r)*DD + nt1*16 + col] = a01[r] + bb1;
            Cg[(rb1+r)*DD + nt0*16 + col] = a10[r] + bb0;
            Cg[(rb1+r)*DD + nt1*16 + col] = a11[r] + bb1;
        }
    }
}

// ---------- K3: fused waves -> relu(@w3+b3)@w4+b4, *mk, atomic-max agg ----------
__global__ __launch_bounds__(256) void feat_kernel(const float* __restrict__ pred_boxes,
                                                   const bf16* __restrict__ w3p, const float* __restrict__ b3,
                                                   const bf16* __restrict__ w4p, const float* __restrict__ b4,
                                                   const float* __restrict__ tv, const int* __restrict__ idx,
                                                   const float* __restrict__ mkA,
                                                   float* __restrict__ agg) {
    __shared__ bf16  A[MB*LDF2];     // 32x576 feature panel (bf16)
    __shared__ bf16  X[MB*LDA2];     // 32x256 intermediate (bf16)
    __shared__ float ldl[MB][4];
    __shared__ float ovs[MB];
    __shared__ float mks[MB];
    __shared__ int   bns[MB];

    const int blkm = blockIdx.x;                 // 0..4999
    const int tid = threadIdx.x, lane = tid & 31, wv = tid >> 5;
    const int h = lane >> 4, col = lane & 15;
    const int nt0 = wv*2, nt1 = nt0 + 1;
    const int r0f = (lane & 15)*LDF2;
    const int r1f = (16 + (lane & 15))*LDF2;
    const int r0x = (lane & 15)*LDA2;
    const int r1x = (16 + (lane & 15))*LDA2;
    const int rb0 = h*8, rb1 = 16 + h*8;

    if (tid < MB) {
        int r = blkm*MB + tid;                   // global (b,n,k) row; r == (b*NN+n)*KK + k
        int bn = r / KK;
        int b = bn / NN;
        int n = bn % NN;
        bns[tid] = bn;
        int j = idx[r];
        float m = mkA[r];
        mks[tid] = m;
        ovs[tid] = tv[r] * m;
        const float* pb = pred_boxes + (size_t)b*NN*4;
#pragma unroll
        for (int c = 0; c < 4; c++) {
            float dc = pb[j*4+c] - pb[n*4+c];
            ldl[tid][c] = logf(fmaxf(fabsf(dc), 1e-7f));
        }
    }
    __syncthreads();

    // build feature panel: [0,64)=overs, [64,576)=sin/cos waves
    const float TWO_PI = 6.28318530717958647692f;
    const float L2_1E4 = 13.28771237954945f;     // log2(10000)
    for (int t = tid; t < MB*F3; t += 256) {
        int rr = t / F3, f = t % F3;
        float val;
        if (f < 64) {
            val = ovs[rr];
        } else {
            int q  = f - 64;
            int i2 = q >> 8;            // coord pair
            int jj = (q >> 7) & 1;      // 0=sin, 1=cos
            int tt = q & 127;
            float dimt = exp2f(L2_1E4 * (float)(2*(tt>>1)) * (1.0f/128.0f));
            float pm = TWO_PI * ldl[rr][2*i2 + jj] / dimt;
            val = jj ? __cosf(pm) : __sinf(pm);
        }
        A[rr*LDF2 + f] = (bf16)val;
    }
    __syncthreads();

    // GEMM1: X = relu(A @ w3 + b3)   (K = 576, 18 k-steps)
    v8f a00 = {}, a01 = {}, a10 = {}, a11 = {};
    for (int kt = 0; kt < F3/32; kt++) {
        v16bf fa0 = load_afrag(&A[r0f], kt*32, h);
        v16bf fa1 = load_afrag(&A[r1f], kt*32, h);
        v16bf fb0 = load_bfrag(w3p, kt, nt0, lane);
        v16bf fb1 = load_bfrag(w3p, kt, nt1, lane);
        a00 = wmma_bf16(fa0, fb0, a00);
        a10 = wmma_bf16(fa1, fb0, a10);
        a01 = wmma_bf16(fa0, fb1, a01);
        a11 = wmma_bf16(fa1, fb1, a11);
    }
    {
        float bb0 = b3[nt0*16 + col], bb1 = b3[nt1*16 + col];
#pragma unroll
        for (int r = 0; r < 8; r++) {
            X[(rb0+r)*LDA2 + nt0*16 + col] = (bf16)fmaxf(a00[r] + bb0, 0.0f);
            X[(rb0+r)*LDA2 + nt1*16 + col] = (bf16)fmaxf(a01[r] + bb1, 0.0f);
            X[(rb1+r)*LDA2 + nt0*16 + col] = (bf16)fmaxf(a10[r] + bb0, 0.0f);
            X[(rb1+r)*LDA2 + nt1*16 + col] = (bf16)fmaxf(a11[r] + bb1, 0.0f);
        }
    }
    __syncthreads();

    // GEMM2: D = X @ w4 + b4; *mk; atomic-max into agg
    a00 = (v8f){}; a01 = (v8f){}; a10 = (v8f){}; a11 = (v8f){};
    for (int kt = 0; kt < DD/32; kt++) {
        v16bf fa0 = load_afrag(&X[r0x], kt*32, h);
        v16bf fa1 = load_afrag(&X[r1x], kt*32, h);
        v16bf fb0 = load_bfrag(w4p, kt, nt0, lane);
        v16bf fb1 = load_bfrag(w4p, kt, nt1, lane);
        a00 = wmma_bf16(fa0, fb0, a00);
        a10 = wmma_bf16(fa1, fb0, a10);
        a01 = wmma_bf16(fa0, fb1, a01);
        a11 = wmma_bf16(fa1, fb1, a11);
    }
    {
        float bb0 = b4[nt0*16 + col], bb1 = b4[nt1*16 + col];
#pragma unroll
        for (int r = 0; r < 8; r++) {
            int ra = rb0 + r, rbq = rb1 + r;
            atomicMaxFloat(&agg[(size_t)bns[ra]*DD + nt0*16 + col], (a00[r] + bb0) * mks[ra]);
            atomicMaxFloat(&agg[(size_t)bns[ra]*DD + nt1*16 + col], (a01[r] + bb1) * mks[ra]);
            atomicMaxFloat(&agg[(size_t)bns[rbq]*DD + nt0*16 + col], (a10[r] + bb0) * mks[rbq]);
            atomicMaxFloat(&agg[(size_t)bns[rbq]*DD + nt1*16 + col], (a11[r] + bb1) * mks[rbq]);
        }
    }
}

// ---------- K4: out = relu((cur*neg + agg)@w5 + b5) * neg ----------
__global__ __launch_bounds__(256) void out_kernel(const float* __restrict__ cur,
                                                  const float* __restrict__ agg,
                                                  const float* __restrict__ seed_mask,
                                                  const bf16* __restrict__ w5p, const float* __restrict__ b5,
                                                  float* __restrict__ outp) {
    __shared__ bf16  A[MB*LDA2];
    __shared__ float nm[MB];
    const int blk = blockIdx.x;                  // 0..499
    const int tid = threadIdx.x, lane = tid & 31, wv = tid >> 5;
    const int h = lane >> 4, col = lane & 15;
    const int nt0 = wv*2, nt1 = nt0 + 1;
    const int r0 = (lane & 15)*LDA2;
    const int r1 = (16 + (lane & 15))*LDA2;
    const int rb0 = h*8, rb1 = 16 + h*8;

    if (tid < MB) nm[tid] = 1.0f - seed_mask[blk*MB + tid];
    __syncthreads();
    const float* Cg = cur + (size_t)blk*MB*DD;
    const float* Ag = agg + (size_t)blk*MB*DD;
    for (int t = tid; t < MB*DD; t += 256) {
        int rr = t >> 8;
        A[rr*LDA2 + (t&255)] = (bf16)(Cg[t] * nm[rr] + Ag[t]);
    }
    __syncthreads();

    v8f a00 = {}, a01 = {}, a10 = {}, a11 = {};
    for (int kt = 0; kt < DD/32; kt++) {
        v16bf fa0 = load_afrag(&A[r0], kt*32, h);
        v16bf fa1 = load_afrag(&A[r1], kt*32, h);
        v16bf fb0 = load_bfrag(w5p, kt, nt0, lane);
        v16bf fb1 = load_bfrag(w5p, kt, nt1, lane);
        a00 = wmma_bf16(fa0, fb0, a00);
        a10 = wmma_bf16(fa1, fb0, a10);
        a01 = wmma_bf16(fa0, fb1, a01);
        a11 = wmma_bf16(fa1, fb1, a11);
    }
    {
        float bb0 = b5[nt0*16 + col], bb1 = b5[nt1*16 + col];
        float* Og = outp + (size_t)blk*MB*DD;
#pragma unroll
        for (int r = 0; r < 8; r++) {
            Og[(rb0+r)*DD + nt0*16 + col] = fmaxf(a00[r] + bb0, 0.0f) * nm[rb0+r];
            Og[(rb0+r)*DD + nt1*16 + col] = fmaxf(a01[r] + bb1, 0.0f) * nm[rb0+r];
            Og[(rb1+r)*DD + nt0*16 + col] = fmaxf(a10[r] + bb0, 0.0f) * nm[rb1+r];
            Og[(rb1+r)*DD + nt1*16 + col] = fmaxf(a11[r] + bb1, 0.0f) * nm[rb1+r];
        }
    }
}

extern "C" void kernel_launch(void* const* d_in, const int* in_sizes, int n_in,
                              void* d_out, int out_size, void* d_ws, size_t ws_size,
                              hipStream_t stream) {
    const float* tgt        = (const float*)d_in[0];
    const float* seed_mask  = (const float*)d_in[1];
    const float* pred_boxes = (const float*)d_in[2];
    const float* w1 = (const float*)d_in[3];  const float* b1 = (const float*)d_in[4];
    const float* w2 = (const float*)d_in[5];  const float* b2 = (const float*)d_in[6];
    const float* w3 = (const float*)d_in[7];  const float* b3 = (const float*)d_in[8];
    const float* w4 = (const float*)d_in[9];  const float* b4 = (const float*)d_in[10];
    const float* w5 = (const float*)d_in[11]; const float* b5 = (const float*)d_in[12];

    float* out  = (float*)d_out;                    // [cur_tgt 16000*256][attn 16*1000*1000]
    float* attn = out + (size_t)ROWS*DD;

    // workspace layout (all offsets keep >=64B alignment)
    float* agg = (float*)d_ws;                      // 16000*256 f32
    float* cur = agg + (size_t)ROWS*DD;             // 16000*256 f32
    float* tv  = cur + (size_t)ROWS*DD;             // 160000 f32
    float* mk  = tv + MROWS;                        // 160000 f32
    int*   idx = (int*)(mk + MROWS);                // 160000 i32
    bf16*  w3p = (bf16*)(idx + MROWS);              // 576*256 bf16
    bf16*  w1p = w3p + (size_t)F3*DD;               // 256*256 bf16 each
    bf16*  w2p = w1p + (size_t)DD*DD;
    bf16*  w4p = w2p + (size_t)DD*DD;
    bf16*  w5p = w4p + (size_t)DD*DD;

    // pack weights into bf16 WMMA B-fragment layout
    pack_weight_kernel<<<(F3*DD + 255)/256, 256, 0, stream>>>(w3, w3p, F3);
    pack_weight_kernel<<<(DD*DD + 255)/256, 256, 0, stream>>>(w1, w1p, DD);
    pack_weight_kernel<<<(DD*DD + 255)/256, 256, 0, stream>>>(w2, w2p, DD);
    pack_weight_kernel<<<(DD*DD + 255)/256, 256, 0, stream>>>(w4, w4p, DD);
    pack_weight_kernel<<<(DD*DD + 255)/256, 256, 0, stream>>>(w5, w5p, DD);

    topk_iou_kernel<<<ROWS, 32, 0, stream>>>(pred_boxes, seed_mask, attn, tv, idx, mk);
    init_agg_kernel<<<(ROWS*DD + 255)/256, 256, 0, stream>>>(agg, ROWS*DD);
    cur_kernel<<<ROWS/MB, 256, 0, stream>>>(tgt, w1p, b1, w2p, b2, cur);
    feat_kernel<<<MROWS/MB, 256, 0, stream>>>(pred_boxes, w3p, b3, w4p, b4, tv, idx, mk, agg);
    out_kernel<<<ROWS/MB, 256, 0, stream>>>(cur, agg, seed_mask, w5p, b5, out);
}
